// DLinear_1717986918411
// MI455X (gfx1250) — compile-verified, matrix operand order checked by reading, not run on hardware
//
#include <hip/hip_runtime.h>

#define T_DIM   720
#define M_DIM   16384        // 32 * 512
#define WINDOW  25
#define BM      128
#define BN      64
#define KB      32
#define KSTEPS  23           // ceil(720/32), tail zero-padded
#define LDS_LD  40           // 32 + 8 halves padding

typedef __bf16 v16bf __attribute__((ext_vector_type(16)));
typedef float  v8f   __attribute__((ext_vector_type(8)));

struct pack32 { uint4 a, b; };   // 32 bytes == v16bf

__device__ __forceinline__ unsigned short f2bf(float f) {
    unsigned u = __float_as_uint(f);
    u += 0x7FFFu + ((u >> 16) & 1u);           // round-to-nearest-even
    return (unsigned short)(u >> 16);
}
__device__ __forceinline__ float bf2f(unsigned short h) {
    return __uint_as_float(((unsigned)h) << 16);
}

// ---------------------------------------------------------------------------
// Kernel 1: fold causal-window-mean + both linear layers into one matrix C
// C[u,t] = W_s[u,t] + sum_{j=t}^{min(t+24,719)} (W_t[u,j]-W_s[u,j])/min(j+1,25)
// Emit C as split bf16 planes (hi + residual lo) plus fused bias.
// ---------------------------------------------------------------------------
__global__ __launch_bounds__(256)
void fuse_weights_kernel(const float* __restrict__ trend_w,
                         const float* __restrict__ trend_b,
                         const float* __restrict__ seasonal_w,
                         const float* __restrict__ seasonal_b,
                         unsigned short* __restrict__ C_hi,
                         unsigned short* __restrict__ C_lo,
                         float* __restrict__ biasC) {
    int id = blockIdx.x * 256 + threadIdx.x;
    if (id >= T_DIM * T_DIM) return;
    int u = id / T_DIM;
    int t = id - u * T_DIM;

    float c = seasonal_w[(size_t)u * T_DIM + t];
    int jend = min(t + WINDOW - 1, T_DIM - 1);
    for (int j = t; j <= jend; ++j) {
        float inv = (j >= WINDOW - 1) ? (1.0f / (float)WINDOW)
                                      : (1.0f / (float)(j + 1));
        c += (trend_w[(size_t)u * T_DIM + j] - seasonal_w[(size_t)u * T_DIM + j]) * inv;
    }
    unsigned short hi = f2bf(c);
    unsigned short lo = f2bf(c - bf2f(hi));
    C_hi[id] = hi;
    C_lo[id] = lo;
    if (t == 0) biasC[u] = trend_b[u] + seasonal_b[u];
}

// ---------------------------------------------------------------------------
// Kernel 2: out[m,u] = sum_t x[m,t] * C[u,t] + bias[u]
// bf16x2 split-precision GEMM on v_wmma_f32_16x16x32_bf16 (wave32).
// ---------------------------------------------------------------------------
__device__ __forceinline__ v16bf ld_frag(const unsigned short* base, int row, int kb) {
    const unsigned short* p = base + row * LDS_LD + kb;
    pack32 pk;
    pk.a = *(const uint4*)(p);        // K = kb .. kb+7
    pk.b = *(const uint4*)(p + 16);   // K = kb+16 .. kb+23
    return __builtin_bit_cast(v16bf, pk);
}

__device__ __forceinline__ void store_tile(float* __restrict__ out,
                                           const float* __restrict__ biasC,
                                           int mbase, int u, v8f acc) {
    if (u >= T_DIM) return;
    float bv = biasC[u];
#pragma unroll
    for (int r = 0; r < 8; ++r)
        out[(size_t)(mbase + r) * T_DIM + u] = acc[r] + bv;
}

__global__ __launch_bounds__(256)
void dlinear_gemm_kernel(const float* __restrict__ x,
                         const unsigned short* __restrict__ C_hi,
                         const unsigned short* __restrict__ C_lo,
                         const float* __restrict__ biasC,
                         float* __restrict__ out) {
    __shared__ __align__(16) unsigned short xs_hi[BM * LDS_LD];
    __shared__ __align__(16) unsigned short xs_lo[BM * LDS_LD];
    __shared__ __align__(16) unsigned short cs_hi[BN * LDS_LD];
    __shared__ __align__(16) unsigned short cs_lo[BN * LDS_LD];

    const int tid  = threadIdx.x;
    const int lane = tid & 31;
    const int wv   = tid >> 5;          // 8 waves
    const int fr   = lane & 15;         // row-in-fragment
    const int kb   = (lane >> 4) * 8;   // K sub-chunk base (0 or 8)

    v8f acc0 = {0.f,0.f,0.f,0.f,0.f,0.f,0.f,0.f};
    v8f acc1 = acc0, acc2 = acc0, acc3 = acc0;

    // x-stage indices: 2 threads per row, 16 cols each
    const int xr = tid >> 1;
    const int xc0 = (tid & 1) * 16;
    const float* xrow = x + (size_t)(blockIdx.y * BM + xr) * T_DIM;

    // C-stage indices: 4 threads per row, 8 cols each
    const int cr = tid >> 2;
    const int cc0 = (tid & 3) * 8;
    const int cu = blockIdx.x * BN + cr;

    for (int ks = 0; ks < KSTEPS; ++ks) {
        const int k0 = ks * KB;
        if (ks) __syncthreads();

        // ---- stage x tile (f32 -> bf16 hi/lo) ----
#pragma unroll
        for (int q = 0; q < 4; ++q) {
            const int c = xc0 + q * 4;
            float4 v;
            if (k0 + c < T_DIM) v = *(const float4*)(xrow + k0 + c);
            else                v = make_float4(0.f, 0.f, 0.f, 0.f);
            float vv[4] = {v.x, v.y, v.z, v.w};
#pragma unroll
            for (int i = 0; i < 4; ++i) {
                unsigned short h = f2bf(vv[i]);
                xs_hi[xr * LDS_LD + c + i] = h;
                xs_lo[xr * LDS_LD + c + i] = f2bf(vv[i] - bf2f(h));
            }
        }

        // ---- stage C tile (already bf16 planes) ----
        {
            uint4 zh = {0u,0u,0u,0u}, zl = {0u,0u,0u,0u};
            if (cu < T_DIM && (k0 + cc0) < T_DIM) {
                zh = *(const uint4*)(C_hi + (size_t)cu * T_DIM + k0 + cc0);
                zl = *(const uint4*)(C_lo + (size_t)cu * T_DIM + k0 + cc0);
            }
            *(uint4*)(&cs_hi[cr * LDS_LD + cc0]) = zh;
            *(uint4*)(&cs_lo[cr * LDS_LD + cc0]) = zl;
        }

        __syncthreads();

        // ---- compute: A strip = wave's 16 rows, 4 N-subtiles ----
        const v16bf a_hi = ld_frag(xs_hi, wv * 16 + fr, kb);
        const v16bf a_lo = ld_frag(xs_lo, wv * 16 + fr, kb);

#define DL_TILE(J, ACC)                                                         \
        {                                                                       \
            v16bf b_hi = ld_frag(cs_hi, (J) * 16 + fr, kb);                     \
            v16bf b_lo = ld_frag(cs_lo, (J) * 16 + fr, kb);                     \
            ACC = __builtin_amdgcn_wmma_f32_16x16x32_bf16(                      \
                false, a_hi, false, b_hi, (short)0, ACC, false, false);         \
            ACC = __builtin_amdgcn_wmma_f32_16x16x32_bf16(                      \
                false, a_hi, false, b_lo, (short)0, ACC, false, false);         \
            ACC = __builtin_amdgcn_wmma_f32_16x16x32_bf16(                      \
                false, a_lo, false, b_hi, (short)0, ACC, false, false);         \
        }
        DL_TILE(0, acc0)
        DL_TILE(1, acc1)
        DL_TILE(2, acc2)
        DL_TILE(3, acc3)
#undef DL_TILE
    }

    // ---- epilogue: D layout -> global, add fused bias ----
    const int mbase = blockIdx.y * BM + wv * 16 + ((lane >> 4) << 3);
    const int ubase = blockIdx.x * BN + fr;
    store_tile(out, biasC, mbase, ubase +  0, acc0);
    store_tile(out, biasC, mbase, ubase + 16, acc1);
    store_tile(out, biasC, mbase, ubase + 32, acc2);
    store_tile(out, biasC, mbase, ubase + 48, acc3);
}

// ---------------------------------------------------------------------------
extern "C" void kernel_launch(void* const* d_in, const int* in_sizes, int n_in,
                              void* d_out, int out_size, void* d_ws, size_t ws_size,
                              hipStream_t stream) {
    (void)in_sizes; (void)n_in; (void)out_size; (void)ws_size;
    const float* x          = (const float*)d_in[0];
    const float* trend_w    = (const float*)d_in[1];
    const float* trend_b    = (const float*)d_in[2];
    const float* seasonal_w = (const float*)d_in[3];
    const float* seasonal_b = (const float*)d_in[4];
    float* out = (float*)d_out;

    unsigned short* C_hi = (unsigned short*)d_ws;
    unsigned short* C_lo = C_hi + (size_t)T_DIM * T_DIM;
    float* biasC = (float*)(C_lo + (size_t)T_DIM * T_DIM);

    fuse_weights_kernel<<<(T_DIM * T_DIM + 255) / 256, 256, 0, stream>>>(
        trend_w, trend_b, seasonal_w, seasonal_b, C_hi, C_lo, biasC);

    dim3 grid((T_DIM + BN - 1) / BN, M_DIM / BM);   // 12 x 128
    dlinear_gemm_kernel<<<grid, 256, 0, stream>>>(x, C_hi, C_lo, biasC, out);
}
